// modified_Net2_73735998538052
// MI455X (gfx1250) — matrix operand order repaired and checked by reading, not compiled, not run on hardware
//
#include <hip/hip_runtime.h>
#include <hip/hip_bf16.h>

#define BATCH 4
#define NP1   1024
#define NP2   1152
#define KNN   20
#define SEL   512
#define RPTS  128

#define CDIV(a,b) (((a)+(b)-1)/(b))

typedef __attribute__((ext_vector_type(16))) __bf16 v16bf;
typedef __attribute__((ext_vector_type(8)))  float  v8f;
typedef __attribute__((ext_vector_type(4)))  __bf16 bf16x4;

#define GF_RELU 1
#define GF_ACC  2
#define GF_TA   4
#define GF_TB   8

// ---------------------------------------------------------------------------
// Generic batched GEMM: C = alpha*(A@B) [+bias] [+C] [relu], bf16 WMMA compute
// Block: 256 threads (8 waves), tile 128(M) x 64(N), K-step 32.
// Fast-path staging (full tiles): b128 global loads + b64 LDS stores.
// ---------------------------------------------------------------------------
__global__ __launch_bounds__(256) void gemm_bf16_kernel(
    const float* __restrict__ A, const float* __restrict__ Bm,
    const float* __restrict__ bias, float* __restrict__ C,
    int M, int N, int K, int lda, int ldb, int ldc,
    long long sA, long long sB, long long sC, float alpha, int flags)
{
    A  += (long long)blockIdx.z * sA;
    Bm += (long long)blockIdx.z * sB;
    C  += (long long)blockIdx.z * sC;

    __shared__ __bf16 As[128 * 32];   // [m][k]
    __shared__ __bf16 Bs[64 * 32];    // [n][k] (transposed stage)

    const int tid  = threadIdx.x;
    const int m_base = blockIdx.x * 128;
    const int n_base = blockIdx.y * 64;
    const int wave = tid >> 5;
    const int lane = tid & 31;
    const int wm = wave & 3;          // 4 waves along M
    const int wn = wave >> 2;         // 2 waves along N
    const int half = lane >> 4;
    const int lr   = lane & 15;

    const bool tileM = (m_base + 128 <= M);
    const bool tileN = (n_base + 64 <= N);

    v8f acc[2][2];
#pragma unroll
    for (int i = 0; i < 2; ++i)
#pragma unroll
        for (int j = 0; j < 2; ++j)
#pragma unroll
            for (int r = 0; r < 8; ++r) acc[i][j][r] = 0.0f;

    for (int k0 = 0; k0 < K; k0 += 32) {
        const bool tileK = (k0 + 32 <= K);

        // ---- stage A tile (f32 -> bf16) ----
        if (!(flags & GF_TA) && tileM && tileK) {
            // fast path: 16B vector loads along K, 8B LDS stores
            const int m0 = tid >> 3;          // 0..31
            const int kq = (tid & 7) * 4;     // 0..28
#pragma unroll
            for (int rep = 0; rep < 4; ++rep) {
                const int mm = m0 + rep * 32;
                const float* src = &A[(long long)(m_base + mm) * lda + (k0 + kq)];
                if (k0 + 32 < K) __builtin_prefetch(src + 32, 0, 1);
                float4 v = *(const float4*)src;
                bf16x4 pk;
                pk[0] = (__bf16)v.x; pk[1] = (__bf16)v.y;
                pk[2] = (__bf16)v.z; pk[3] = (__bf16)v.w;
                *(bf16x4*)&As[mm * 32 + kq] = pk;
            }
        } else {
            for (int i = tid; i < 128 * 32; i += 256) {
                int m = i >> 5, k = i & 31;
                int gm = m_base + m, gk = k0 + k;
                float v = 0.0f;
                if (gm < M && gk < K)
                    v = (flags & GF_TA) ? A[(long long)gk * lda + gm]
                                        : A[(long long)gm * lda + gk];
                As[m * 32 + k] = (__bf16)v;
            }
        }

        // ---- stage B tile into [n][k] ----
        if (tileN && tileK) {
            if (flags & GF_TB) {
                // contiguous along K: vector loads + b64 LDS stores
                const int n0 = tid >> 2;          // 0..63
                const int kq = (tid & 3) * 8;     // 0,8,16,24
                const float* src = &Bm[(long long)(n_base + n0) * ldb + (k0 + kq)];
                float4 v0 = *(const float4*)src;
                float4 v1 = *(const float4*)(src + 4);
                bf16x4 p0, p1;
                p0[0] = (__bf16)v0.x; p0[1] = (__bf16)v0.y;
                p0[2] = (__bf16)v0.z; p0[3] = (__bf16)v0.w;
                p1[0] = (__bf16)v1.x; p1[1] = (__bf16)v1.y;
                p1[2] = (__bf16)v1.z; p1[3] = (__bf16)v1.w;
                *(bf16x4*)&Bs[n0 * 32 + kq]     = p0;
                *(bf16x4*)&Bs[n0 * 32 + kq + 4] = p1;
            } else {
                // [k][n] source: coalesced along N, straight-line b16 LDS stores
                const int n0 = tid & 63;
                const int kb = tid >> 6;          // 0..3
#pragma unroll
                for (int rep = 0; rep < 8; ++rep) {
                    const int k = kb + rep * 4;
                    float v = Bm[(long long)(k0 + k) * ldb + (n_base + n0)];
                    Bs[n0 * 32 + k] = (__bf16)v;
                }
            }
        } else {
            for (int i = tid; i < 64 * 32; i += 256) {
                int n = i >> 5, k = i & 31;
                int gn = n_base + n, gk = k0 + k;
                float v = 0.0f;
                if (gn < N && gk < K)
                    v = (flags & GF_TB) ? Bm[(long long)gn * ldb + gk]
                                        : Bm[(long long)gk * ldb + gn];
                Bs[n * 32 + k] = (__bf16)v;
            }
        }
        __syncthreads();

        v16bf afrag[2], bfrag[2];
#pragma unroll
        for (int fm = 0; fm < 2; ++fm) {
            const __bf16* src = &As[(wm * 32 + fm * 16 + lr) * 32];
#pragma unroll
            for (int v = 0; v < 8; ++v) {
                int kk = ((v & 4) ? 16 : 0) + half * 8 + (v & 3) * 2;
                afrag[fm][2 * v]     = src[kk];
                afrag[fm][2 * v + 1] = src[kk + 1];
            }
        }
#pragma unroll
        for (int fn = 0; fn < 2; ++fn) {
            const __bf16* src = &Bs[(wn * 32 + fn * 16 + lr) * 32];
#pragma unroll
            for (int v = 0; v < 8; ++v) {
                int kk = ((v & 4) ? 16 : 0) + half * 8 + (v & 3) * 2;
                bfrag[fn][2 * v]     = src[kk];
                bfrag[fn][2 * v + 1] = src[kk + 1];
            }
        }
#pragma unroll
        for (int fm = 0; fm < 2; ++fm)
#pragma unroll
            for (int fn = 0; fn < 2; ++fn)
                acc[fm][fn] = __builtin_amdgcn_wmma_f32_16x16x32_bf16(
                    false, afrag[fm], false, bfrag[fn],
                    (short)0, acc[fm][fn], false, false);
        __syncthreads();
    }

#pragma unroll
    for (int fm = 0; fm < 2; ++fm)
#pragma unroll
        for (int fn = 0; fn < 2; ++fn)
#pragma unroll
            for (int r = 0; r < 8; ++r) {
                int gm = m_base + wm * 32 + fm * 16 + half * 8 + r;
                int gn = n_base + wn * 32 + fn * 16 + lr;
                if (gm < M && gn < N) {
                    float v = acc[fm][fn][r] * alpha;
                    if (bias) v += bias[gn];
                    if (flags & GF_ACC) v += C[(long long)gm * ldc + gn];
                    if (flags & GF_RELU) v = fmaxf(v, 0.0f);
                    C[(long long)gm * ldc + gn] = v;
                }
            }
}

static void gemm(hipStream_t st, const float* A, const float* B, const float* bias,
                 float* C, int M, int N, int K, int lda, int ldb, int ldc,
                 long long sA, long long sB, long long sC, int batch,
                 float alpha, int flags)
{
    dim3 grid(CDIV(M, 128), CDIV(N, 64), batch);
    gemm_bf16_kernel<<<grid, 256, 0, st>>>(A, B, bias, C, M, N, K, lda, ldb, ldc,
                                           sA, sB, sC, alpha, flags);
}

// ---------------------------------------------------------------------------
// Block reductions (blockDim must equal BD)
// ---------------------------------------------------------------------------
template<int BD> __device__ inline float blk_max(float v) {
    __shared__ float sh[BD];
    int t = threadIdx.x; sh[t] = v; __syncthreads();
    for (int s = BD >> 1; s > 0; s >>= 1) {
        if (t < s) sh[t] = fmaxf(sh[t], sh[t + s]);
        __syncthreads();
    }
    float r = sh[0]; __syncthreads(); return r;
}
template<int BD> __device__ inline float blk_sum(float v) {
    __shared__ float sh[BD];
    int t = threadIdx.x; sh[t] = v; __syncthreads();
    for (int s = BD >> 1; s > 0; s >>= 1) {
        if (t < s) sh[t] += sh[t + s];
        __syncthreads();
    }
    float r = sh[0]; __syncthreads(); return r;
}

// ---------------------------------------------------------------------------
// KNN: per-point insertion sort of 20 smallest sq-distances (index tie-break)
// ---------------------------------------------------------------------------
__global__ void knn_kernel(const float* __restrict__ P, int Np, int* __restrict__ idx)
{
    int t = blockIdx.x * blockDim.x + threadIdx.x;
    if (t >= BATCH * Np) return;
    int b = t / Np, i = t % Np;
    const float* Pb = P + (long long)b * Np * 6;
    float xi = Pb[i * 6], yi = Pb[i * 6 + 1], zi = Pb[i * 6 + 2];
    float bd[KNN]; int bi[KNN];
    for (int q = 0; q < KNN; ++q) { bd[q] = 3.0e38f; bi[q] = 0; }
    for (int j = 0; j < Np; ++j) {
        float dx = Pb[j * 6] - xi, dy = Pb[j * 6 + 1] - yi, dz = Pb[j * 6 + 2] - zi;
        float d = dx * dx + dy * dy + dz * dz;
        if (d < bd[KNN - 1]) {
            int p = KNN - 1;
            while (p > 0 && bd[p - 1] > d) { bd[p] = bd[p - 1]; bi[p] = bi[p - 1]; --p; }
            bd[p] = d; bi[p] = j;
        }
    }
    for (int q = 0; q < KNN; ++q) idx[(long long)t * KNN + q] = bi[q];
}

__global__ void edge_kernel(const float* __restrict__ P, const int* __restrict__ idx,
                            int Np, int b, float* __restrict__ E)
{
    int t = blockIdx.x * blockDim.x + threadIdx.x;
    if (t >= Np * KNN) return;
    int i = t / KNN, q = t % KNN;
    int j = idx[((long long)b * Np + i) * KNN + q];
    const float* Pb = P + (long long)b * Np * 6;
    const float* pi = Pb + i * 6;
    const float* pj = Pb + j * 6;
    float* e = E + (long long)t * 6;
    e[0] = pj[0] - pi[0]; e[1] = pj[1] - pi[1]; e[2] = pj[2] - pi[2];
    e[3] = pi[0]; e[4] = pi[1]; e[5] = pi[2];
}

__global__ void maxk_kernel(const float* __restrict__ H, int Np, int b, float* __restrict__ F)
{
    int t = blockIdx.x * blockDim.x + threadIdx.x;
    if (t >= Np * 256) return;
    int i = t >> 8, c = t & 255;
    float m = -3.0e38f;
    for (int q = 0; q < KNN; ++q)
        m = fmaxf(m, H[((long long)i * KNN + q) * 256 + c]);
    F[((long long)b * Np + i) * 256 + c] = m;
}

// column sums (optionally |.|) of dense (rows x cols) matrices per batch
__global__ void colsum_kernel(const float* __restrict__ X, int rows, int cols,
                              long long strideB, int total, int do_abs,
                              float* __restrict__ out)
{
    int t = blockIdx.x * blockDim.x + threadIdx.x;
    if (t >= total) return;
    int b = t / cols, j = t % cols;
    const float* Xb = X + (long long)b * strideB;
    float s = 0.0f;
    for (int i = 0; i < rows; ++i) {
        float v = Xb[(long long)i * cols + j];
        s += do_abs ? fabsf(v) : v;
    }
    out[t] = s;
}

__global__ void adjnorm_kernel(const float* __restrict__ Ain, const float* __restrict__ cs,
                               int Np, long long total, float* __restrict__ An)
{
    long long t = (long long)blockIdx.x * blockDim.x + threadIdx.x;
    if (t >= total) return;
    int b = (int)(t / ((long long)Np * Np));
    int j = (int)(t % Np);
    An[t] = Ain[t] / fmaxf(cs[b * Np + j], 1e-12f);
}

__global__ void instnorm_reduce_kernel(const float* __restrict__ X, long long cnt,
                                       float* __restrict__ stats)
{
    int b = blockIdx.x, tid = threadIdx.x;
    const float* Xb = X + (long long)b * cnt;
    float s = 0.0f, ss = 0.0f;
    for (long long i = tid; i < cnt; i += 256) { float v = Xb[i]; s += v; ss += v * v; }
    __shared__ float s1[256], s2[256];
    s1[tid] = s; s2[tid] = ss; __syncthreads();
    for (int st = 128; st > 0; st >>= 1) {
        if (tid < st) { s1[tid] += s1[tid + st]; s2[tid] += s2[tid + st]; }
        __syncthreads();
    }
    if (tid == 0) {
        float m = s1[0] / (float)cnt;
        stats[b * 2] = m;
        stats[b * 2 + 1] = s2[0] / (float)cnt - m * m;
    }
}

__global__ void instnorm_apply_kernel(float* __restrict__ X, long long cnt,
                                      const float* __restrict__ stats,
                                      const float* __restrict__ w,
                                      const float* __restrict__ bb, long long total)
{
    long long t = (long long)blockIdx.x * blockDim.x + threadIdx.x;
    if (t >= total) return;
    int b = (int)(t / cnt);
    float m = stats[b * 2], var = stats[b * 2 + 1];
    X[t] = (X[t] - m) * rsqrtf(var + 1e-5f) * w[0] + bb[0];
}

__global__ void pad_kernel(const float* __restrict__ S, float* __restrict__ LA,
                           int R, int C, long long total)
{
    long long t = (long long)blockIdx.x * blockDim.x + threadIdx.x;
    if (t >= total) return;
    int cf = C + 1, rf = R + 1;
    int b = (int)(t / ((long long)rf * cf));
    long long rem = t % ((long long)rf * cf);
    int i = (int)(rem / cf), j = (int)(rem % cf);
    LA[t] = (i < R && j < C) ? S[((long long)b * R + i) * C + j] : 0.0f;
}

__global__ void sk_row_kernel(float* __restrict__ LA, int R, int C)
{
    int i = blockIdx.x, b = blockIdx.y, cf = C + 1;
    float* row = LA + ((long long)b * (R + 1) + i) * cf;
    float m = -3.0e38f;
    for (int j = threadIdx.x; j < cf; j += 128) m = fmaxf(m, row[j]);
    m = blk_max<128>(m);
    float s = 0.0f;
    for (int j = threadIdx.x; j < cf; j += 128) s += __expf(row[j] - m);
    s = blk_sum<128>(s);
    float lse = m + __logf(s);
    for (int j = threadIdx.x; j < cf; j += 128) row[j] -= lse;
}

__global__ void sk_col_kernel(float* __restrict__ LA, int R, int C)
{
    int j = blockIdx.x, b = blockIdx.y, cf = C + 1, rf = R + 1;
    float* col = LA + (long long)b * rf * cf + j;
    float m = -3.0e38f;
    for (int i = threadIdx.x; i < rf; i += 128) m = fmaxf(m, col[(long long)i * cf]);
    m = blk_max<128>(m);
    float s = 0.0f;
    for (int i = threadIdx.x; i < rf; i += 128) s += __expf(col[(long long)i * cf] - m);
    s = blk_sum<128>(s);
    float lse = m + __logf(s);
    for (int i = threadIdx.x; i < rf; i += 128) col[(long long)i * cf] -= lse;
}

__global__ void exp_kernel(const float* __restrict__ LA, float* __restrict__ Sout,
                           int R, int C, long long total)
{
    long long t = (long long)blockIdx.x * blockDim.x + threadIdx.x;
    if (t >= total) return;
    int b = (int)(t / ((long long)R * C));
    long long rem = t % ((long long)R * C);
    int i = (int)(rem / C), j = (int)(rem % C);
    Sout[t] = __expf(LA[((long long)b * (R + 1) + i) * (C + 1) + j]);
}

__global__ void rowsum_kernel(const float* __restrict__ X, int rows_total, int cols,
                              float* __restrict__ out)
{
    int t = blockIdx.x * blockDim.x + threadIdx.x;
    if (t >= rows_total) return;
    const float* row = X + (long long)t * cols;
    float s = 0.0f;
    for (int j = 0; j < cols; ++j) s += row[j];
    out[t] = s;
}

__global__ void topk_rank_kernel(const float* __restrict__ vals, int Nv, int Ktop,
                                 int largest, int* __restrict__ out)
{
    __shared__ float sv[1152];
    int b = blockIdx.x;
    const float* v = vals + (long long)b * Nv;
    for (int i = threadIdx.x; i < Nv; i += blockDim.x) sv[i] = v[i];
    __syncthreads();
    for (int e = threadIdx.x; e < Nv; e += blockDim.x) {
        float ve = sv[e]; int rank = 0;
        for (int j = 0; j < Nv; ++j) {
            float vj = sv[j];
            bool before = largest ? (vj > ve) : (vj < ve);
            before = before || (vj == ve && j < e);
            rank += before ? 1 : 0;
        }
        if (rank < Ktop) out[(long long)b * Ktop + rank] = e;
    }
}

__global__ void gather_rows_kernel(const float* __restrict__ src, const int* __restrict__ idx,
                                   int rows_out, int W, long long strideSrcB,
                                   float* __restrict__ dst, long long total)
{
    long long t = (long long)blockIdx.x * blockDim.x + threadIdx.x;
    if (t >= total) return;
    long long rw = (long long)rows_out * W;
    int b = (int)(t / rw);
    long long rem = t % rw;
    int r = (int)(rem / W), c = (int)(rem % W);
    dst[t] = src[(long long)b * strideSrcB + (long long)idx[b * rows_out + r] * W + c];
}

// softmax over axis=1 (rows): column stats (max, sumexp)
__global__ void axis1_stats_kernel(const float* __restrict__ X, int Nrows, int D,
                                   float* __restrict__ cmax, float* __restrict__ csum)
{
    int t = blockIdx.x * blockDim.x + threadIdx.x;
    if (t >= BATCH * D) return;
    int b = t / D, d = t % D;
    const float* base = X + (long long)b * Nrows * D + d;
    float m = -3.0e38f;
    for (int i = 0; i < Nrows; ++i) m = fmaxf(m, base[(long long)i * D]);
    float s = 0.0f;
    for (int i = 0; i < Nrows; ++i) s += __expf(base[(long long)i * D] - m);
    cmax[t] = m; csum[t] = s;
}

__global__ void softmax_score_kernel(const float* __restrict__ X,
                                     const float* __restrict__ cmax,
                                     const float* __restrict__ csum,
                                     int Nrows, int D, float* __restrict__ out)
{
    int t = blockIdx.x * blockDim.x + threadIdx.x;
    if (t >= BATCH * Nrows) return;
    int b = t / Nrows, n = t % Nrows;
    const float* row = X + ((long long)b * Nrows + n) * D;
    float s = 0.0f;
    for (int d = 0; d < D; ++d)
        s += __expf(row[d] - cmax[b * D + d]) / csum[b * D + d];
    out[t] = s;
}

__global__ void row_softmax_kernel(float* __restrict__ X, int N)
{
    float* row = X + (long long)blockIdx.x * N;
    float m = -3.0e38f;
    for (int j = threadIdx.x; j < N; j += 128) m = fmaxf(m, row[j]);
    m = blk_max<128>(m);
    float s = 0.0f;
    for (int j = threadIdx.x; j < N; j += 128) s += __expf(row[j] - m);
    s = blk_sum<128>(s);
    float inv = 1.0f / s;
    for (int j = threadIdx.x; j < N; j += 128) row[j] = __expf(row[j] - m) * inv;
}

// ---------------------------------------------------------------------------
extern "C" void kernel_launch(void* const* d_in, const int* in_sizes, int n_in,
                              void* d_out, int out_size, void* d_ws, size_t ws_size,
                              hipStream_t stream)
{
    (void)in_sizes; (void)n_in; (void)out_size; (void)ws_size;
    const float* P_src = (const float*)d_in[0];
    const float* P_tgt = (const float*)d_in[1];
    const float* A_src = (const float*)d_in[2];
    const float* A_tgt = (const float*)d_in[3];
    const float* dgW0 = (const float*)d_in[4];
    const float* dgW1 = (const float*)d_in[5];
    const float* dgW2 = (const float*)d_in[6];
    const float* dgW3 = (const float*)d_in[7];
    const float* dgb0 = (const float*)d_in[8];
    const float* dgb1 = (const float*)d_in[9];
    const float* dgb2 = (const float*)d_in[10];
    const float* dgb3 = (const float*)d_in[11];
    const float* dg_pW = (const float*)d_in[12];
    const float* dg_pb = (const float*)d_in[13];
    const float* dg_eW = (const float*)d_in[14];
    const float* dg_eb = (const float*)d_in[15];
    const float* g0aW = (const float*)d_in[16];
    const float* g0ab = (const float*)d_in[17];
    const float* g0uW = (const float*)d_in[18];
    const float* g0ub = (const float*)d_in[19];
    const float* g1aW = (const float*)d_in[20];
    const float* g1ab = (const float*)d_in[21];
    const float* g1uW = (const float*)d_in[22];
    const float* g1ub = (const float*)d_in[23];
    const float* aff0 = (const float*)d_in[24];
    const float* aff1 = (const float*)d_in[25];
    const float* in0w = (const float*)d_in[26];
    const float* in0b = (const float*)d_in[27];
    const float* in1w = (const float*)d_in[28];
    const float* in1b = (const float*)d_in[29];
    const float* cc0W = (const float*)d_in[30];
    const float* cc0b = (const float*)d_in[31];
    // d_in[32], d_in[33] = cc1_W/cc1_b: unused in the forward pass
    const float* TWq = (const float*)d_in[34];
    const float* TWk = (const float*)d_in[35];
    const float* TWv = (const float*)d_in[36];
    const float* TWo = (const float*)d_in[37];

    // ---- workspace bump allocator ----
    char* wsb = (char*)d_ws;
    size_t off = 0;
    auto alloc = [&](size_t nbytes) -> void* {
        void* p = wsb + off;
        off += (nbytes + 255) & ~(size_t)255;
        return p;
    };
    auto fall = [&](size_t nfloats) -> float* { return (float*)alloc(nfloats * 4); };
    auto iall = [&](size_t nints) -> int* { return (int*)alloc(nints * 4); };

    int*   IDXS = iall((size_t)BATCH * NP1 * KNN);
    int*   IDXT = iall((size_t)BATCH * NP2 * KNN);
    float* FS   = fall((size_t)BATCH * NP1 * 256);
    float* FT   = fall((size_t)BATCH * NP2 * 256);
    float* EMBSF= fall((size_t)BATCH * NP1 * 1024);
    float* EMBTF= fall((size_t)BATCH * NP2 * 1024);
    float* X1   = fall((size_t)BATCH * NP1 * 512);
    float* X2   = fall((size_t)BATCH * NP2 * 512);
    float* AN   = fall((size_t)BATCH * NP2 * NP2);
    float* Hb   = fall((size_t)BATCH * NP2 * 512);
    float* T1   = fall((size_t)BATCH * NP1 * 512);
    float* LA   = fall((size_t)BATCH * (NP1 + 1) * (NP2 + 1));
    float* Sm   = fall((size_t)BATCH * NP1 * NP2);
    float* SX2  = fall((size_t)BATCH * NP1 * 512);
    float* STX1 = fall((size_t)BATCH * NP2 * 512);
    float* E1   = fall((size_t)BATCH * NP1 * 512);
    float* E2   = fall((size_t)BATCH * NP2 * 512);
    float* E2R  = fall((size_t)BATCH * NP1 * 512);
    float* ES   = fall((size_t)BATCH * SEL * 512);
    float* ET   = fall((size_t)BATCH * SEL * 512);
    float* Qb   = fall((size_t)BATCH * SEL * 512);
    float* Kb   = fall((size_t)BATCH * SEL * 512);
    float* Vb   = fall((size_t)BATCH * SEL * 512);
    float* ATTb = fall((size_t)BATCH * SEL * SEL);
    float* AVb  = fall((size_t)BATCH * SEL * 512);
    float* SEb  = fall((size_t)BATCH * SEL * 512);
    float* TEb  = fall((size_t)BATCH * SEL * 512);
    float* ASb  = fall((size_t)BATCH * SEL * SEL);
    float* ATn  = fall((size_t)BATCH * SEL * SEL);
    float* Y1   = fall((size_t)BATCH * SEL * 256);
    float* Y2   = fall((size_t)BATCH * SEL * 256);
    float* S2R  = fall((size_t)BATCH * SEL * SEL);
    float* LA2  = fall((size_t)BATCH * (SEL + 1) * (SEL + 1));
    float* EE   = fall((size_t)NP2 * KNN * 6);
    float* EH1  = fall((size_t)NP2 * KNN * 256);
    float* EH2  = fall((size_t)NP2 * KNN * 256);
    float* CS   = fall((size_t)BATCH * NP2);
    float* CS2  = fall((size_t)BATCH * NP2);
    float* STATS= fall((size_t)BATCH * 2);
    float* CMAX = fall((size_t)BATCH * 512);
    float* CSUM = fall((size_t)BATCH * 512);
    int*   RIDX = iall((size_t)BATCH * NP1);
    int*   MIDX = iall((size_t)BATCH * RPTS);
    int*   SIDX = iall((size_t)BATCH * SEL);
    int*   TIDX = iall((size_t)BATCH * SEL);

    float* out_s2  = (float*)d_out;
    float* out_src = out_s2 + (size_t)BATCH * SEL * SEL;
    float* out_ref = out_src + (size_t)BATCH * SEL;
    float* out_mp  = out_ref + (size_t)BATCH * SEL;

    const float invD = 0.044194173824159216f; // 1/sqrt(512)

    // ---- Phase 0: KNN ----
    knn_kernel<<<CDIV(BATCH * NP1, 128), 128, 0, stream>>>(P_src, NP1, IDXS);
    knn_kernel<<<CDIV(BATCH * NP2, 128), 128, 0, stream>>>(P_tgt, NP2, IDXT);

    // ---- Phase 1: DGCNN edge MLP (per cloud, per batch chunk) ----
    for (int cl = 0; cl < 2; ++cl) {
        const float* P  = cl ? P_tgt : P_src;
        const int*   ix = cl ? IDXT : IDXS;
        float*       F  = cl ? FT : FS;
        int          Np = cl ? NP2 : NP1;
        for (int b = 0; b < BATCH; ++b) {
            int M = Np * KNN;
            edge_kernel<<<CDIV(M, 256), 256, 0, stream>>>(P, ix, Np, b, EE);
            gemm(stream, EE,  dgW0, dgb0, EH1, M, 64, 6,    6,  64,  64, 0,0,0, 1, 1.f, GF_RELU);
            gemm(stream, EH1, dgW1, dgb1, EH2, M, 64, 64,  64,  64,  64, 0,0,0, 1, 1.f, GF_RELU);
            gemm(stream, EH2, dgW2, dgb2, EH1, M, 128, 64, 64, 128, 128, 0,0,0, 1, 1.f, GF_RELU);
            gemm(stream, EH1, dgW3, dgb3, EH2, M, 256, 128,128, 256, 256, 0,0,0, 1, 1.f, GF_RELU);
            maxk_kernel<<<CDIV(Np * 256, 256), 256, 0, stream>>>(EH2, Np, b, F);
        }
    }
    gemm(stream, FS, dg_pW, dg_pb, EMBSF,       BATCH*NP1, 512, 256, 256, 512, 1024, 0,0,0, 1, 1.f, GF_RELU);
    gemm(stream, FS, dg_eW, dg_eb, EMBSF + 512, BATCH*NP1, 512, 256, 256, 512, 1024, 0,0,0, 1, 1.f, GF_RELU);
    gemm(stream, FT, dg_pW, dg_pb, EMBTF,       BATCH*NP2, 512, 256, 256, 512, 1024, 0,0,0, 1, 1.f, GF_RELU);
    gemm(stream, FT, dg_eW, dg_eb, EMBTF + 512, BATCH*NP2, 512, 256, 256, 512, 1024, 0,0,0, 1, 1.f, GF_RELU);

    // ---- Phase 2: gconv0 ----
    colsum_kernel<<<CDIV(BATCH*NP1,256),256,0,stream>>>(A_src, NP1, NP1, (long long)NP1*NP1, BATCH*NP1, 1, CS);
    adjnorm_kernel<<<CDIV(BATCH*NP1*NP1,256),256,0,stream>>>(A_src, CS, NP1, (long long)BATCH*NP1*NP1, AN);
    gemm(stream, EMBSF, g0aW, g0ab, Hb, BATCH*NP1, 512, 1024, 1024, 512, 512, 0,0,0, 1, 1.f, GF_RELU);
    gemm(stream, EMBSF, g0uW, g0ub, X1, BATCH*NP1, 512, 1024, 1024, 512, 512, 0,0,0, 1, 1.f, GF_RELU);
    gemm(stream, AN, Hb, nullptr, X1, NP1, 512, NP1, NP1, 512, 512,
         (long long)NP1*NP1, (long long)NP1*512, (long long)NP1*512, BATCH, 1.f, GF_ACC);

    colsum_kernel<<<CDIV(BATCH*NP2,256),256,0,stream>>>(A_tgt, NP2, NP2, (long long)NP2*NP2, BATCH*NP2, 1, CS);
    adjnorm_kernel<<<CDIV(BATCH*NP2*NP2,256),256,0,stream>>>(A_tgt, CS, NP2, (long long)BATCH*NP2*NP2, AN);
    gemm(stream, EMBTF, g0aW, g0ab, Hb, BATCH*NP2, 512, 1024, 1024, 512, 512, 0,0,0, 1, 1.f, GF_RELU);
    gemm(stream, EMBTF, g0uW, g0ub, X2, BATCH*NP2, 512, 1024, 1024, 512, 512, 0,0,0, 1, 1.f, GF_RELU);
    gemm(stream, AN, Hb, nullptr, X2, NP2, 512, NP2, NP2, 512, 512,
         (long long)NP2*NP2, (long long)NP2*512, (long long)NP2*512, BATCH, 1.f, GF_ACC);

    // ---- Phase 3: affinity + inst-norm + Sinkhorn ----
    gemm(stream, X1, aff0, nullptr, T1, BATCH*NP1, 512, 512, 512, 512, 512, 0,0,0, 1, 1.f, 0);
    gemm(stream, T1, X2, nullptr, Sm, NP1, NP2, 512, 512, 512, NP2,
         (long long)NP1*512, (long long)NP2*512, (long long)NP1*NP2, BATCH, 1.f, GF_TB);
    instnorm_reduce_kernel<<<BATCH, 256, 0, stream>>>(Sm, (long long)NP1*NP2, STATS);
    instnorm_apply_kernel<<<CDIV(BATCH*NP1*NP2,256),256,0,stream>>>(Sm, (long long)NP1*NP2, STATS, in0w, in0b, (long long)BATCH*NP1*NP2);
    pad_kernel<<<CDIV(BATCH*(NP1+1)*(NP2+1),256),256,0,stream>>>(Sm, LA, NP1, NP2, (long long)BATCH*(NP1+1)*(NP2+1));
    for (int it = 0; it < 20; ++it) {
        sk_row_kernel<<<dim3(NP1, BATCH), 128, 0, stream>>>(LA, NP1, NP2);
        sk_col_kernel<<<dim3(NP2, BATCH), 128, 0, stream>>>(LA, NP1, NP2);
    }
    exp_kernel<<<CDIV(BATCH*NP1*NP2,256),256,0,stream>>>(LA, Sm, NP1, NP2, (long long)BATCH*NP1*NP2);

    // ---- Phase 4: cross-concat projections ----
    gemm(stream, Sm, X2, nullptr, SX2, NP1, 512, NP2, NP2, 512, 512,
         (long long)NP1*NP2, (long long)NP2*512, (long long)NP1*512, BATCH, 1.f, 0);
    gemm(stream, Sm, X1, nullptr, STX1, NP2, 512, NP1, NP2, 512, 512,
         (long long)NP1*NP2, (long long)NP1*512, (long long)NP2*512, BATCH, 1.f, GF_TA);
    gemm(stream, X1,  cc0W,            cc0b,    E1, BATCH*NP1, 512, 512, 512, 512, 512, 0,0,0, 1, 1.f, 0);
    gemm(stream, SX2, cc0W + 512*512,  nullptr, E1, BATCH*NP1, 512, 512, 512, 512, 512, 0,0,0, 1, 1.f, GF_ACC);
    gemm(stream, X2,  cc0W,            cc0b,    E2, BATCH*NP2, 512, 512, 512, 512, 512, 0,0,0, 1, 1.f, 0);
    gemm(stream, STX1,cc0W + 512*512,  nullptr, E2, BATCH*NP2, 512, 512, 512, 512, 512, 0,0,0, 1, 1.f, GF_ACC);

    // ---- Phase 5: top-k selections & gathers ----
    colsum_kernel<<<CDIV(BATCH*NP2,256),256,0,stream>>>(Sm, NP1, NP2, (long long)NP1*NP2, BATCH*NP2, 0, CS);
    topk_rank_kernel<<<BATCH, 256, 0, stream>>>(CS, NP2, NP1, 1, RIDX);
    topk_rank_kernel<<<BATCH, 256, 0, stream>>>(CS, NP2, RPTS, 0, MIDX);
    gather_rows_kernel<<<CDIV(BATCH*RPTS*6,256),256,0,stream>>>(P_tgt, MIDX, RPTS, 6, (long long)NP2*6, out_mp, (long long)BATCH*RPTS*6);

    axis1_stats_kernel<<<CDIV(BATCH*512,256),256,0,stream>>>(E1, NP1, 512, CMAX, CSUM);
    softmax_score_kernel<<<CDIV(BATCH*NP1,256),256,0,stream>>>(E1, CMAX, CSUM, NP1, 512, CS2);
    topk_rank_kernel<<<BATCH, 256, 0, stream>>>(CS2, NP1, SEL, 1, SIDX);
    gather_rows_kernel<<<CDIV(BATCH*SEL*512,256),256,0,stream>>>(E1, SIDX, SEL, 512, (long long)NP1*512, ES, (long long)BATCH*SEL*512);

    gather_rows_kernel<<<CDIV(BATCH*NP1*512,256),256,0,stream>>>(E2, RIDX, NP1, 512, (long long)NP2*512, E2R, (long long)BATCH*NP1*512);
    axis1_stats_kernel<<<CDIV(BATCH*512,256),256,0,stream>>>(E2R, NP1, 512, CMAX, CSUM);
    softmax_score_kernel<<<CDIV(BATCH*NP1,256),256,0,stream>>>(E2R, CMAX, CSUM, NP1, 512, CS2);
    topk_rank_kernel<<<BATCH, 256, 0, stream>>>(CS2, NP1, SEL, 1, TIDX);
    gather_rows_kernel<<<CDIV(BATCH*SEL*512,256),256,0,stream>>>(E2, TIDX, SEL, 512, (long long)NP2*512, ET, (long long)BATCH*SEL*512);

    // ---- Phase 6: cross attention (se, te) ----
    size_t selbytes = sizeof(float) * (size_t)BATCH * SEL * 512;
    hipMemcpyAsync(SEb, ES, selbytes, hipMemcpyDeviceToDevice, stream);
    gemm(stream, ES, TWq, nullptr, Qb, BATCH*SEL, 512, 512, 512, 512, 512, 0,0,0, 1, 1.f, 0);
    gemm(stream, ET, TWk, nullptr, Kb, BATCH*SEL, 512, 512, 512, 512, 512, 0,0,0, 1, 1.f, 0);
    gemm(stream, ET, TWv, nullptr, Vb, BATCH*SEL, 512, 512, 512, 512, 512, 0,0,0, 1, 1.f, 0);
    gemm(stream, Qb, Kb, nullptr, ATTb, SEL, SEL, 512, 512, 512, SEL,
         (long long)SEL*512, (long long)SEL*512, (long long)SEL*SEL, BATCH, invD, GF_TB);
    row_softmax_kernel<<<BATCH*SEL, 128, 0, stream>>>(ATTb, SEL);
    gemm(stream, ATTb, Vb, nullptr, AVb, SEL, 512, SEL, SEL, 512, 512,
         (long long)SEL*SEL, (long long)SEL*512, (long long)SEL*512, BATCH, 1.f, 0);
    gemm(stream, AVb, TWo, nullptr, SEb, BATCH*SEL, 512, 512, 512, 512, 512, 0,0,0, 1, 1.f, GF_ACC);

    hipMemcpyAsync(TEb, ET, selbytes, hipMemcpyDeviceToDevice, stream);
    gemm(stream, ET, TWq, nullptr, Qb, BATCH*SEL, 512, 512, 512, 512, 512, 0,0,0, 1, 1.f, 0);
    gemm(stream, ES, TWk, nullptr, Kb, BATCH*SEL, 512, 512, 512, 512, 512, 0,0,0, 1, 1.f, 0);
    gemm(stream, ES, TWv, nullptr, Vb, BATCH*SEL, 512, 512, 512, 512, 512, 0,0,0, 1, 1.f, 0);
    gemm(stream, Qb, Kb, nullptr, ATTb, SEL, SEL, 512, 512, 512, SEL,
         (long long)SEL*512, (long long)SEL*512, (long long)SEL*SEL, BATCH, invD, GF_TB);
    row_softmax_kernel<<<BATCH*SEL, 128, 0, stream>>>(ATTb, SEL);
    gemm(stream, ATTb, Vb, nullptr, AVb, SEL, 512, SEL, SEL, 512, 512,
         (long long)SEL*SEL, (long long)SEL*512, (long long)SEL*512, BATCH, 1.f, 0);
    gemm(stream, AVb, TWo, nullptr, TEb, BATCH*SEL, 512, 512, 512, 512, 512, 0,0,0, 1, 1.f, GF_ACC);

    // ---- Phase 7: self-similarity graphs + gconv1 ----
    gemm(stream, SEb, SEb, nullptr, ASb, SEL, SEL, 512, 512, 512, SEL,
         (long long)SEL*512, (long long)SEL*512, (long long)SEL*SEL, BATCH, invD, GF_TB);
    row_softmax_kernel<<<BATCH*SEL, 128, 0, stream>>>(ASb, SEL);
    gemm(stream, TEb, TEb, nullptr, ATn, SEL, SEL, 512, 512, 512, SEL,
         (long long)SEL*512, (long long)SEL*512, (long long)SEL*SEL, BATCH, invD, GF_TB);
    row_softmax_kernel<<<BATCH*SEL, 128, 0, stream>>>(ATn, SEL);

    colsum_kernel<<<CDIV(BATCH*SEL,256),256,0,stream>>>(ASb, SEL, SEL, (long long)SEL*SEL, BATCH*SEL, 1, CS);
    adjnorm_kernel<<<CDIV(BATCH*SEL*SEL,256),256,0,stream>>>(ASb, CS, SEL, (long long)BATCH*SEL*SEL, ASb);
    gemm(stream, ES, g1aW, g1ab, Hb, BATCH*SEL, 256, 512, 512, 256, 256, 0,0,0, 1, 1.f, GF_RELU);
    gemm(stream, ES, g1uW, g1ub, Y1, BATCH*SEL, 256, 512, 512, 256, 256, 0,0,0, 1, 1.f, GF_RELU);
    gemm(stream, ASb, Hb, nullptr, Y1, SEL, 256, SEL, SEL, 256, 256,
         (long long)SEL*SEL, (long long)SEL*256, (long long)SEL*256, BATCH, 1.f, GF_ACC);

    colsum_kernel<<<CDIV(BATCH*SEL,256),256,0,stream>>>(ATn, SEL, SEL, (long long)SEL*SEL, BATCH*SEL, 1, CS);
    adjnorm_kernel<<<CDIV(BATCH*SEL*SEL,256),256,0,stream>>>(ATn, CS, SEL, (long long)BATCH*SEL*SEL, ATn);
    gemm(stream, ET, g1aW, g1ab, Hb, BATCH*SEL, 256, 512, 512, 256, 256, 0,0,0, 1, 1.f, GF_RELU);
    gemm(stream, ET, g1uW, g1ub, Y2, BATCH*SEL, 256, 512, 512, 256, 256, 0,0,0, 1, 1.f, GF_RELU);
    gemm(stream, ATn, Hb, nullptr, Y2, SEL, 256, SEL, SEL, 256, 256,
         (long long)SEL*SEL, (long long)SEL*256, (long long)SEL*256, BATCH, 1.f, GF_ACC);

    // ---- Phase 8: final affinity + Sinkhorn + outputs ----
    gemm(stream, Y1, aff1, nullptr, T1, BATCH*SEL, 256, 256, 256, 256, 256, 0,0,0, 1, 1.f, 0);
    gemm(stream, T1, Y2, nullptr, S2R, SEL, SEL, 256, 256, 256, SEL,
         (long long)SEL*256, (long long)SEL*256, (long long)SEL*SEL, BATCH, 1.f, GF_TB);
    instnorm_reduce_kernel<<<BATCH, 256, 0, stream>>>(S2R, (long long)SEL*SEL, STATS);
    instnorm_apply_kernel<<<CDIV(BATCH*SEL*SEL,256),256,0,stream>>>(S2R, (long long)SEL*SEL, STATS, in1w, in1b, (long long)BATCH*SEL*SEL);
    pad_kernel<<<CDIV(BATCH*(SEL+1)*(SEL+1),256),256,0,stream>>>(S2R, LA2, SEL, SEL, (long long)BATCH*(SEL+1)*(SEL+1));
    for (int it = 0; it < 20; ++it) {
        sk_row_kernel<<<dim3(SEL, BATCH), 128, 0, stream>>>(LA2, SEL, SEL);
        sk_col_kernel<<<dim3(SEL, BATCH), 128, 0, stream>>>(LA2, SEL, SEL);
    }
    exp_kernel<<<CDIV(BATCH*SEL*SEL,256),256,0,stream>>>(LA2, out_s2, SEL, SEL, (long long)BATCH*SEL*SEL);
    rowsum_kernel<<<CDIV(BATCH*SEL,256),256,0,stream>>>(out_s2, BATCH*SEL, SEL, out_src);
    colsum_kernel<<<CDIV(BATCH*SEL,256),256,0,stream>>>(out_s2, SEL, SEL, (long long)SEL*SEL, BATCH*SEL, 0, out_ref);
}
